// DCM_50414326120808
// MI455X (gfx1250) — matrix operand order repaired and checked by reading, not compiled, not run on hardware
//
#include <hip/hip_runtime.h>
#include <hip/hip_bf16.h>

// ---------------------------------------------------------------------------
// Problem constants
// ---------------------------------------------------------------------------
#define B_  16
#define C_  256
#define H_  96
#define W_  96
#define BN_EPS 1e-5f
#define LDSW 264   // padded w-stride (elements) of the staged cin-major tile

typedef __attribute__((ext_vector_type(8)))  float  v8f;
typedef __attribute__((ext_vector_type(8)))  __bf16 v8bf;
typedef __attribute__((ext_vector_type(16))) __bf16 v16bf;
typedef __attribute__((ext_vector_type(2)))  __bf16 v2bf;
typedef unsigned int ui4 __attribute__((ext_vector_type(4)));
typedef int v4i __attribute__((vector_size(16)));   // matches builtin param type

#if defined(__HIP_DEVICE_COMPILE__) && __has_builtin(__builtin_amdgcn_global_load_async_to_lds_b128)
#  define HAVE_ASYNC_LDS 1
#else
#  define HAVE_ASYNC_LDS 0
#endif

__device__ __forceinline__ v8f wmma_bf16(v16bf A, v16bf Bm, v8f Cacc) {
#if defined(__HIP_DEVICE_COMPILE__)
    return __builtin_amdgcn_wmma_f32_16x16x32_bf16(
        /*neg_a=*/false, A, /*neg_b=*/false, Bm,
        /*c_mod=*/(short)0, Cacc, /*reuse_a=*/false, /*reuse_b=*/false);
#else
    (void)A; (void)Bm;
    return Cacc;
#endif
}

__device__ __forceinline__ v16bf cat8(v8bf lo, v8bf hi) {
    return __builtin_shufflevector(lo, hi, 0,1,2,3,4,5,6,7,8,9,10,11,12,13,14,15);
}

// 16-byte global -> LDS stage (async if available)
__device__ __forceinline__ void stage16(const __bf16* g, __bf16* l) {
#if HAVE_ASYNC_LDS
    // addrspacecast to AS1/AS3, then reinterpret pointee as v4i
    const __attribute__((address_space(1))) __bf16* g1 =
        (const __attribute__((address_space(1))) __bf16*)g;
    __attribute__((address_space(3))) __bf16* l3 =
        (__attribute__((address_space(3))) __bf16*)l;
    __builtin_amdgcn_global_load_async_to_lds_b128(
        (__attribute__((address_space(1))) v4i*)g1,
        (__attribute__((address_space(3))) v4i*)l3,
        /*offset=*/0, /*cpol=*/0);
#else
    *(ui4*)l = *(const ui4*)g;
#endif
}

__device__ __forceinline__ void wait_stage() {
#if HAVE_ASYNC_LDS
#  if __has_builtin(__builtin_amdgcn_s_wait_asynccnt)
    __builtin_amdgcn_s_wait_asynccnt(0);
#  else
    asm volatile("s_wait_asynccnt 0" ::: "memory");
#  endif
#endif
}

// ---------------------------------------------------------------------------
// 1) adaptive_avg_pool2d(x, 3): one block per (b,c); 9 bins of 32x32
// ---------------------------------------------------------------------------
__global__ __launch_bounds__(256) void pool_k(const float* __restrict__ x,
                                              float* __restrict__ pooled) {
    __shared__ float red[256];
    const int bc  = blockIdx.x;
    const int tid = threadIdx.x;
    const float* xp = x + (size_t)bc * (H_ * W_);
    for (int bin = 0; bin < 9; ++bin) {
        const int bi = bin / 3, bj = bin % 3;
        float v = 0.f;
        for (int k = tid; k < 1024; k += 256) {
            const int r = k >> 5, cc = k & 31;
            v += xp[(bi * 32 + r) * W_ + bj * 32 + cc];
        }
        red[tid] = v;
        __syncthreads();
        for (int s = 128; s > 0; s >>= 1) {
            if (tid < s) red[tid] += red[tid + s];
            __syncthreads();
        }
        if (tid == 0) pooled[(size_t)bc * 9 + bin] = red[0] * (1.f / 1024.f);
        __syncthreads();
    }
}

// ---------------------------------------------------------------------------
// 2) gen[b,o,p] = sum_c w_gen[o,c] * pooled[b,c,p] + b_gen[o]
// ---------------------------------------------------------------------------
__global__ __launch_bounds__(256) void gen_k(const float* __restrict__ pooled,
                                             const float* __restrict__ wg,
                                             const float* __restrict__ bg,
                                             float* __restrict__ gen) {
    __shared__ float pc[256];
    const int blk = blockIdx.x;
    const int p = blk % 9, b = blk / 9;
    const int o = threadIdx.x;
    pc[o] = pooled[((size_t)(b * C_ + o)) * 9 + p];
    __syncthreads();
    float s = bg[o];
    const float* w = wg + (size_t)o * C_;
    for (int c = 0; c < C_; ++c) s = fmaf(w[c], pc[c], s);
    gen[((size_t)(b * C_ + o)) * 9 + p] = s;
}

// ---------------------------------------------------------------------------
// 3) Pack w_c1 (fp32 OIHW) -> bf16 WMMA-A layout:
//    wA[((tap*8 + cinBlk)*256 + o)*32 + k]
// ---------------------------------------------------------------------------
__global__ __launch_bounds__(256) void packw_k(const float* __restrict__ w,
                                               __bf16* __restrict__ wA) {
    const int idx = blockIdx.x * 256 + threadIdx.x;
    if (idx >= 9 * 8 * 256 * 32) return;
    const int k   = idx & 31;
    const int o   = (idx >> 5) & 255;
    const int cb  = (idx >> 13) & 7;
    const int tap = idx >> 16;
    const int r = tap / 3, s = tap % 3;
    wA[idx] = (__bf16)w[(((size_t)o * C_ + cb * 32 + k) * 3 + r) * 3 + s];
}

// ---------------------------------------------------------------------------
// 4) fp32 NCHW -> bf16 NHWC (one block per (b,h) row; LDS transpose)
// ---------------------------------------------------------------------------
__global__ __launch_bounds__(256) void nhwc_k(const float* __restrict__ in,
                                              __bf16* __restrict__ out) {
    __shared__ __bf16 t[C_ * W_];                  // [c][w]  48 KB
    const int bh  = blockIdx.x;                    // b*96 + h
    const int tid = threadIdx.x;
    const float* ip = in + (size_t)(bh / H_) * C_ * H_ * W_ + (size_t)(bh % H_) * W_;
    for (int idx = tid; idx < C_ * W_; idx += 256) {
        const int w = idx % W_, c = idx / W_;      // coalesced along w
        t[c * W_ + w] = (__bf16)ip[(size_t)c * (H_ * W_) + w];
    }
    __syncthreads();
    __bf16* op = out + (size_t)bh * W_ * C_;
    for (int idx = tid; idx < W_ * (C_ / 2); idx += 256) {
        const int cp = idx & 127, w = idx >> 7;    // coalesced along c
        const int c = cp * 2;
        v2bf pr = { t[c * W_ + w], t[(c + 1) * W_ + w] };
        *(v2bf*)(op + (size_t)w * C_ + c) = pr;
    }
}

// ---------------------------------------------------------------------------
// 5) zero BN accumulators
// ---------------------------------------------------------------------------
__global__ void zero_k(float* __restrict__ p, int n) {
    const int i = blockIdx.x * blockDim.x + threadIdx.x;
    if (i < n) p[i] = 0.f;
}

// ---------------------------------------------------------------------------
// 6) conv3x3 (pad 1) implicit GEMM, V_WMMA_F32_16X16X32_BF16.
//    Block: 8 waves; tile = 256 Cout x 32 pixels (fixed b,h).
//    Full 3x34x256-cin halo staged once in LDS (async global->LDS if avail).
//    Wave w: Cout [w*32, w*32+32) as two 16-row A tiles; 4 WMMAs / (cb,tap).
// ---------------------------------------------------------------------------
__global__ __launch_bounds__(256) void conv_wmma_k(const __bf16* __restrict__ nhwc,
                                                   const __bf16* __restrict__ wA,
                                                   const float*  __restrict__ bias,
                                                   float* __restrict__ y,
                                                   float* __restrict__ sum,
                                                   float* __restrict__ sumsq) {
    __shared__ __bf16 smem[3 * 34 * LDSW];         // [r][w(34)][cin(256)] padded

    const int bx    = blockIdx.x;                  // wtile + 3*h
    const int wtile = bx % 3;
    const int h     = bx / 3;
    const int b     = blockIdx.y;
    const int tid   = threadIdx.x;
    const int wave  = tid >> 5;
    const int lane  = tid & 31;
    const int m     = lane & 15;
    const int hi    = lane >> 4;
    const int cwave = wave * 32;
    const int wbase = wtile * 32;

    // ---- stage halo tile: 3 rows x 34 w x 256 cin, 16-B chunks ----
    const ui4 zero4 = {0, 0, 0, 0};
    for (int chunk = tid; chunk < 3 * 34 * 32; chunk += 256) {
        const int rw = chunk >> 5, part = chunk & 31;    // 32 x 16B = 512B/pixel
        const int r = rw / 34, w = rw - r * 34;
        const int gh = h - 1 + r, gw = wbase - 1 + w;
        const int loff = rw * LDSW + part * 8;           // elements
        if ((unsigned)gh < (unsigned)H_ && (unsigned)gw < (unsigned)W_) {
            const __bf16* gsrc =
                nhwc + ((((size_t)b * H_ + gh) * W_ + gw) * C_ + part * 8);
            stage16(gsrc, &smem[loff]);
        } else {
            *(ui4*)(&smem[loff]) = zero4;
        }
    }
    wait_stage();
    __syncthreads();

    v8f acc0 = {}, acc1 = {}, acc2 = {}, acc3 = {};

    for (int cb = 0; cb < 8; ++cb) {
        #pragma unroll
        for (int tap = 0; tap < 9; ++tap) {
            const int r = tap / 3, s = tap % 3;
            // A: lanes 0-15 K{0-15}; lanes 16-31 K{16-31}
            const __bf16* wp =
                wA + (((size_t)tap * 8 + cb) * 256 + cwave + m) * 32;
            const v16bf A0 = cat8(*(const v8bf*)(wp + 16 * hi),
                                  *(const v8bf*)(wp + 8 + 16 * hi));
            const v16bf A1 = cat8(*(const v8bf*)(wp + 512 + 16 * hi),
                                  *(const v8bf*)(wp + 512 + 8 + 16 * hi));
            // B: column = lane&15, K-half = lane>>4
            const __bf16* lp = &smem[(r * 34 + s) * LDSW + cb * 32 + 16 * hi];
            const v8bf* q0 = (const v8bf*)(lp + (size_t)m * LDSW);
            const v8bf* q1 = (const v8bf*)(lp + (size_t)(16 + m) * LDSW);
            const v16bf B0 = cat8(q0[0], q0[1]);
            const v16bf B1 = cat8(q1[0], q1[1]);
            acc0 = wmma_bf16(A0, B0, acc0);
            acc1 = wmma_bf16(A0, B1, acc1);
            acc2 = wmma_bf16(A1, B0, acc2);
            acc3 = wmma_bf16(A1, B1, acc3);
        }
    }

    // Epilogue: D layout -> cout = cwave + e + 8*hi (+16 for A1); px = m | m+16
    const int px0 = wbase + m;
    const int px1 = wbase + 16 + m;
    #pragma unroll
    for (int e = 0; e < 8; ++e) {
        const int co0 = cwave + e + 8 * hi;
        const int co1 = co0 + 16;
        const float bv0 = bias[co0];
        const float bv1 = bias[co1];
        const float a0 = acc0[e] + bv0, a1 = acc1[e] + bv0;
        const float a2 = acc2[e] + bv1, a3 = acc3[e] + bv1;
        const size_t base0 = (((size_t)b * C_ + co0) * H_ + h) * W_;
        const size_t base1 = (((size_t)b * C_ + co1) * H_ + h) * W_;
        y[base0 + px0] = a0;
        y[base0 + px1] = a1;
        y[base1 + px0] = a2;
        y[base1 + px1] = a3;
        float s1 = a0 + a1, s2 = a0 * a0 + a1 * a1;
        float t1 = a2 + a3, t2 = a2 * a2 + a3 * a3;
        #pragma unroll
        for (int off = 1; off < 16; off <<= 1) {   // reduce the 16 pixel lanes
            s1 += __shfl_xor(s1, off, 32);
            s2 += __shfl_xor(s2, off, 32);
            t1 += __shfl_xor(t1, off, 32);
            t2 += __shfl_xor(t2, off, 32);
        }
        if (m == 0) {
            atomicAdd(&sum[co0], s1);
            atomicAdd(&sumsq[co0], s2);
            atomicAdd(&sum[co1], t1);
            atomicAdd(&sumsq[co1], t2);
        }
    }
}

// ---------------------------------------------------------------------------
// 7) BN finalize
// ---------------------------------------------------------------------------
__global__ void bnfin_k(const float* __restrict__ sum,
                        const float* __restrict__ sumsq,
                        float* __restrict__ mean, float* __restrict__ rstd) {
    const int c = threadIdx.x;
    const float invN = 1.f / ((float)B_ * H_ * W_);
    const float mu  = sum[c] * invN;
    const float var = sumsq[c] * invN - mu * mu;
    mean[c] = mu;
    rstd[c] = rsqrtf(var + BN_EPS);
}

// ---------------------------------------------------------------------------
// 8) fused BN + ReLU + dynamic per-(b,c) depthwise 3x3 conv (pad 1)
// ---------------------------------------------------------------------------
__global__ __launch_bounds__(256) void dw_k(const float* __restrict__ y,
                                            const float* __restrict__ gen,
                                            const float* __restrict__ mean,
                                            const float* __restrict__ rstd,
                                            const float* __restrict__ gamma,
                                            const float* __restrict__ beta,
                                            float* __restrict__ out) {
    __shared__ float tile[34 * 98];
    const int bz    = blockIdx.x;
    const int htile = (bz % 3) * 32;
    const int bc    = bz / 3;                      // b*C + c
    const int c     = bc % C_;
    const int tid   = threadIdx.x;

    const float sA = gamma[c] * rstd[c];
    const float sB = beta[c] - sA * mean[c];

    float g[9];
    const float* gp = gen + (size_t)bc * 9;
    #pragma unroll
    for (int t = 0; t < 9; ++t) g[t] = gp[t];

    const float* yp = y + (size_t)bc * (H_ * W_);
    for (int idx = tid; idx < 34 * 98; idx += 256) {
        const int r = idx / 98, w = idx % 98;
        const int gh = htile - 1 + r, gw = w - 1;
        float v = 0.f;
        if ((unsigned)gh < (unsigned)H_ && (unsigned)gw < (unsigned)W_)
            v = fmaxf(fmaf(yp[gh * W_ + gw], sA, sB), 0.f);
        tile[idx] = v;
    }
    __syncthreads();

    for (int o = tid; o < 32 * W_; o += 256) {
        const int orow = o / W_, ocol = o % W_;
        float a = 0.f;
        #pragma unroll
        for (int i = 0; i < 3; ++i)
            #pragma unroll
            for (int j = 0; j < 3; ++j)
                a = fmaf(g[i * 3 + j], tile[(orow + i) * 98 + ocol + j], a);
        out[(size_t)bc * (H_ * W_) + (htile + orow) * W_ + ocol] = a;
    }
}

// ---------------------------------------------------------------------------
// launch
// ---------------------------------------------------------------------------
extern "C" void kernel_launch(void* const* d_in, const int* in_sizes, int n_in,
                              void* d_out, int out_size, void* d_ws, size_t ws_size,
                              hipStream_t stream) {
    const float* x      = (const float*)d_in[0];
    const float* conv   = (const float*)d_in[1];
    const float* w_gen  = (const float*)d_in[2];
    const float* b_gen  = (const float*)d_in[3];
    const float* w_c1   = (const float*)d_in[4];
    const float* b_c1   = (const float*)d_in[5];
    const float* gamma  = (const float*)d_in[6];
    const float* beta   = (const float*)d_in[7];
    float* out = (float*)d_out;

    // workspace layout (bytes)
    char* wsb = (char*)d_ws;
    float*  pooled = (float*)(wsb + 0);            //    147,456 B
    float*  gen    = (float*)(wsb + 147456);       //    147,456 B
    float*  sum    = (float*)(wsb + 294912);       //      1,024 B
    float*  sumsq  = (float*)(wsb + 295936);       //      1,024 B
    float*  mean   = (float*)(wsb + 296960);       //      1,024 B
    float*  rstd   = (float*)(wsb + 297984);       //      1,024 B
    __bf16* wA     = (__bf16*)(wsb + 299008);      //  1,179,648 B
    __bf16* nhwc   = (__bf16*)(wsb + 1478656);     // 75,497,472 B
    float*  yraw   = (float*)(wsb + 76976128);     // 150,994,944 B -> 227,971,072

    pool_k <<<B_ * C_, 256, 0, stream>>>(x, pooled);
    gen_k  <<<B_ * 9, 256, 0, stream>>>(pooled, w_gen, b_gen, gen);
    packw_k<<<(9 * 8 * 256 * 32 + 255) / 256, 256, 0, stream>>>(w_c1, wA);
    nhwc_k <<<B_ * H_, 256, 0, stream>>>(conv, nhwc);
    zero_k <<<2, 256, 0, stream>>>(sum, 512);

    conv_wmma_k<<<dim3(3 * H_, B_), 256, 0, stream>>>(nhwc, wA, b_c1, yraw,
                                                      sum, sumsq);
    bnfin_k<<<1, 256, 0, stream>>>(sum, sumsq, mean, rstd);
    dw_k   <<<B_ * C_ * 3, 256, 0, stream>>>(yraw, gen, mean, rstd,
                                             gamma, beta, out);
    (void)in_sizes; (void)n_in; (void)out_size; (void)ws_size;
}
